// FlowMatchingPosterior_24532853194800
// MI455X (gfx1250) — matrix operand order repaired and checked by reading, not compiled, not run on hardware
//
#include <hip/hip_runtime.h>
#include <math.h>

typedef __bf16 bf16;
typedef __attribute__((ext_vector_type(16))) __bf16 v16bf;
typedef __attribute__((ext_vector_type(8)))  __bf16 v8bf;
typedef __attribute__((ext_vector_type(8)))  float  v8f;

#define BQ 8192
#define FQ 128
#define CQ 256
#define HQ 512
#define LQ 4
#define SQ 10

// ---------------------------------------------------------------------------
// JVP GEMM: Y_p = act(A_p @ W + bias), Y_t = act'(pre) * (A_t @ W)
// A: [M,K] bf16 row-major (M=8192), W: [K,N] bf16 row-major.
// Workgroup: 256 thr = 8 waves; tile 128(M) x 64(N); each wave 32x32 (2x2
// subtiles) for BOTH primal and tangent -> 8 v_wmma_f32_16x16x32_bf16 / K-step,
// branch-free (HAS_T / FLAGS are template params). LDS is double-buffered so
// the next tile's global_load_b128s overlap the current tile's WMMAs.
// FLAGS: 1 = exact-erf GELU epilogue, 2 = write f32 outs, 4 = write bf16 outs.
// ---------------------------------------------------------------------------
template <bool HAS_T, int FLAGS>
__global__ __launch_bounds__(256)
void jvp_gemm(const bf16* __restrict__ Ap, const bf16* __restrict__ At,
              const bf16* __restrict__ W,  const float* __restrict__ bias,
              int N, int K,
              float* __restrict__ Yp32, float* __restrict__ Yt32,
              bf16* __restrict__ Ypb,  bf16* __restrict__ Ytb)
{
  __shared__ bf16 sAp[2][128 * 40];
  __shared__ bf16 sAt[2][128 * 40];
  __shared__ bf16 sW [2][64 * 40];   // transposed: sW[n][k]

  const int tid  = threadIdx.x;
  const int m0   = blockIdx.y * 128;
  const int n0   = blockIdx.x * 64;
  const int lane = tid & 31;
  const int wv   = tid >> 5;
  const int mb   = (wv & 3) * 32;   // wave M offset in tile
  const int nb   = (wv >> 2) * 32;  // wave N offset in tile
  const int half = lane >> 4;       // lane group 0/1
  const int lrow = lane & 15;

  // staging thread mapping (A: 2 chunks of 8 halves; W: 1 chunk of 8 halves)
  const int arow0 = tid >> 2,        ach0 = tid & 3;
  const int arow1 = (tid + 256) >> 2, ach1 = (tid + 256) & 3;
  const int wkr = tid >> 3, wnc = tid & 7;

  auto stage = [&](int buf, int k0) {
    uint4 a0 = *(const uint4*)(Ap + (size_t)(m0 + arow0) * K + k0 + ach0 * 8);
    uint4 a1 = *(const uint4*)(Ap + (size_t)(m0 + arow1) * K + k0 + ach1 * 8);
    uint4 w  = *(const uint4*)(W  + (size_t)(k0 + wkr) * N + n0 + wnc * 8);
    *(uint4*)&sAp[buf][arow0 * 40 + ach0 * 8] = a0;
    *(uint4*)&sAp[buf][arow1 * 40 + ach1 * 8] = a1;
    if constexpr (HAS_T) {
      uint4 t0 = *(const uint4*)(At + (size_t)(m0 + arow0) * K + k0 + ach0 * 8);
      uint4 t1 = *(const uint4*)(At + (size_t)(m0 + arow1) * K + k0 + ach1 * 8);
      *(uint4*)&sAt[buf][arow0 * 40 + ach0 * 8] = t0;
      *(uint4*)&sAt[buf][arow1 * 40 + ach1 * 8] = t1;
    }
    const bf16* pv = (const bf16*)&w;
#pragma unroll
    for (int j = 0; j < 8; ++j) sW[buf][(wnc * 8 + j) * 40 + wkr] = pv[j];
  };

  v8f accp[2][2] = {};
  v8f acct[2][2] = {};

  stage(0, 0);
  __syncthreads();

  for (int k0 = 0; k0 < K; k0 += 32) {
    const int cur = (k0 >> 5) & 1;
    if (k0 + 32 < K) {
      // gfx1250 global_prefetch_b8: pull next tile's lines ahead of staging
      __builtin_prefetch((const void*)(Ap + (size_t)m0 * K + k0 + 64), 0, 1);
      stage(cur ^ 1, k0 + 32);
    }

    // --- build fragments per documented CDNA5 VGPR layouts ---
    v16bf afp[2], aft[2], bw[2];
#pragma unroll
    for (int ms = 0; ms < 2; ++ms) {
      int r = mb + ms * 16 + lrow;
      *((v8bf*)&afp[ms])     = *(const v8bf*)&sAp[cur][r * 40 + half * 8];
      *((v8bf*)&afp[ms] + 1) = *(const v8bf*)&sAp[cur][r * 40 + 16 + half * 8];
      if constexpr (HAS_T) {
        *((v8bf*)&aft[ms])     = *(const v8bf*)&sAt[cur][r * 40 + half * 8];
        *((v8bf*)&aft[ms] + 1) = *(const v8bf*)&sAt[cur][r * 40 + 16 + half * 8];
      }
    }
#pragma unroll
    for (int ns = 0; ns < 2; ++ns) {
      int c = nb + ns * 16 + lrow;
      *((v8bf*)&bw[ns])     = *(const v8bf*)&sW[cur][c * 40 + half * 8];
      *((v8bf*)&bw[ns] + 1) = *(const v8bf*)&sW[cur][c * 40 + 16 + half * 8];
    }
#pragma unroll
    for (int ns = 0; ns < 2; ++ns)
#pragma unroll
      for (int ms = 0; ms < 2; ++ms) {
        accp[ms][ns] = __builtin_amdgcn_wmma_f32_16x16x32_bf16(
            false, afp[ms], false, bw[ns], (short)0, accp[ms][ns], false, false);
        if constexpr (HAS_T)
          acct[ms][ns] = __builtin_amdgcn_wmma_f32_16x16x32_bf16(
              false, aft[ms], false, bw[ns], (short)0, acct[ms][ns], false, false);
      }
    __syncthreads();
  }

  // --- epilogue: C layout lane=N col, vgpr e -> M = e + 8*half ---
#pragma unroll
  for (int ns = 0; ns < 2; ++ns) {
    int gcol = n0 + nb + ns * 16 + lrow;
    float bcol = bias ? bias[gcol] : 0.f;
#pragma unroll
    for (int ms = 0; ms < 2; ++ms) {
#pragma unroll
      for (int e = 0; e < 8; ++e) {
        int grow = m0 + mb + ms * 16 + e + 8 * half;
        float xp = accp[ms][ns][e] + bcol;
        float ct = HAS_T ? acct[ms][ns][e] : 0.f;
        float yp, yt;
        if constexpr ((FLAGS & 1) != 0) {
          float ph = 0.5f * (1.f + erff(xp * 0.70710678118654752f));
          yp = xp * ph;
          yt = HAS_T
                 ? (ph + xp * expf(-0.5f * xp * xp) * 0.39894228040143268f) * ct
                 : 0.f;
        } else {
          yp = xp; yt = ct;
        }
        size_t o = (size_t)grow * N + gcol;
        if constexpr ((FLAGS & 2) != 0) {
          Yp32[o] = yp;
          if constexpr (HAS_T) Yt32[o] = yt;
        }
        if constexpr ((FLAGS & 4) != 0) {
          Ypb[o] = (bf16)yp;
          if constexpr (HAS_T) Ytb[o] = (bf16)yt;
        }
      }
    }
  }
}

// ---------------------------------------------------------------------------
// Fused residual + LayerNorm (+JVP) (+ optional "+ scale*xatt" post-add).
// One workgroup per row; H = 512, 256 threads x 2 cols.
// ---------------------------------------------------------------------------
__global__ __launch_bounds__(256)
void ln_jvp(float* __restrict__ h, float* __restrict__ ht,
            const float* __restrict__ dp, const float* __restrict__ dtn,
            const float* __restrict__ g, const float* __restrict__ be,
            const float* __restrict__ xatt, const float* __restrict__ scale_ptr,
            int l, bf16* __restrict__ hb, bf16* __restrict__ htb)
{
  __shared__ float r0[256], r1[256], r2[256], r3[256];
  const int row = blockIdx.x, tid = threadIdx.x;
  const size_t o = (size_t)row * HQ;
  float s[2], sd[2], p0 = 0, p1 = 0, p2 = 0, p3 = 0;
#pragma unroll
  for (int i = 0; i < 2; ++i) {
    int c = tid + i * 256;
    s[i]  = h[o + c]  + dp[o + c];
    sd[i] = ht[o + c] + dtn[o + c];
    p0 += s[i]; p1 += s[i] * s[i]; p2 += sd[i]; p3 += s[i] * sd[i];
  }
  r0[tid] = p0; r1[tid] = p1; r2[tid] = p2; r3[tid] = p3;
  __syncthreads();
  for (int off = 128; off > 0; off >>= 1) {
    if (tid < off) {
      r0[tid] += r0[tid + off]; r1[tid] += r1[tid + off];
      r2[tid] += r2[tid + off]; r3[tid] += r3[tid + off];
    }
    __syncthreads();
  }
  const float invH = 1.f / (float)HQ;
  float m   = r0[0] * invH;
  float var = r1[0] * invH - m * m;
  float dm  = r2[0] * invH;
  float c2  = r3[0] * invH - m * dm;        // mean((x-m)*xdot)
  float inv = rsqrtf(var + 1e-5f);
  float sc  = xatt ? scale_ptr[l] : 0.f;
#pragma unroll
  for (int i = 0; i < 2; ++i) {
    int c = tid + i * 256;
    float shat = (s[i] - m) * inv;
    float yp = shat * g[c] + be[c];
    if (xatt) yp += sc * xatt[o + c];
    float yt = g[c] * inv * ((sd[i] - dm) - shat * c2 * inv);
    h[o + c] = yp; ht[o + c] = yt;
    hb[o + c] = (bf16)yp; htb[o + c] = (bf16)yt;
  }
}

// Time embedding (batch-uniform): hb[s] = bi + bt2 + gelu(t_s*Wt1+bt1) @ Wt2
__global__ void temb_kernel(const float* __restrict__ Wt1, const float* __restrict__ bt1,
                            const float* __restrict__ Wt2, const float* __restrict__ bt2,
                            const float* __restrict__ bi, float* __restrict__ hb)
{
  __shared__ float a[HQ];
  const int sI = blockIdx.x, j = threadIdx.x;
  float tv = fminf(fmaxf(0.1f * (float)sI, 0.f), 1.f);
  float x = tv * Wt1[j] + bt1[j];
  a[j] = x * 0.5f * (1.f + erff(x * 0.70710678118654752f));
  __syncthreads();
  float acc = bi[j] + bt2[j];
  for (int k = 0; k < HQ; ++k) acc += a[k] * Wt2[k * HQ + j];
  hb[sI * HQ + j] = acc;
}

__global__ void castf2b(const float* __restrict__ x, bf16* __restrict__ y, int n)
{
  for (int i = blockIdx.x * blockDim.x + threadIdx.x; i < n; i += gridDim.x * blockDim.x)
    y[i] = (bf16)x[i];
}

__global__ void init_zld(const float* __restrict__ x, float* __restrict__ z,
                         float* __restrict__ ld)
{
  int i = blockIdx.x * blockDim.x + threadIdx.x;
  if (i < BQ * FQ) z[i] = x[i];
  if (i < BQ) ld[i] = 0.f;
}

// z += dt*v ; ld += dt * <u,Ju> (wave32 shuffle reduction, 1 wave per row)
__global__ __launch_bounds__(256)
void update_zld(float* __restrict__ z, float* __restrict__ ld,
                const float* __restrict__ v, const float* __restrict__ Ju,
                const float* __restrict__ uk, float dtv)
{
  const int tid = threadIdx.x;
  const int row = blockIdx.x * 8 + (tid >> 5);
  const int lane = tid & 31;
  float s = 0.f;
#pragma unroll
  for (int j = 0; j < 4; ++j) {
    int c = lane + j * 32;
    size_t o = (size_t)row * FQ + c;
    s += uk[o] * Ju[o];
    z[o] += dtv * v[o];
  }
#pragma unroll
  for (int m = 16; m >= 1; m >>= 1) s += __shfl_xor(s, m, 32);
  if (lane == 0) ld[row] += dtv * s;
}

// ---------------------------------------------------------------------------
extern "C" void kernel_launch(void* const* d_in, const int* in_sizes, int n_in,
                              void* d_out, int out_size, void* d_ws, size_t ws_size,
                              hipStream_t stream)
{
  (void)in_sizes; (void)n_in; (void)out_size; (void)ws_size;
  const float* x    = (const float*)d_in[0];
  const float* ctxF = (const float*)d_in[1];
  const float* u    = (const float*)d_in[2];
  const float* Wi   = (const float*)d_in[3];
  const float* bi   = (const float*)d_in[4];
  const float* Wt1  = (const float*)d_in[5];
  const float* bt1  = (const float*)d_in[6];
  const float* Wt2  = (const float*)d_in[7];
  const float* bt2  = (const float*)d_in[8];
  const float* Wc1  = (const float*)d_in[9];
  const float* bc1  = (const float*)d_in[10];
  const float* Wc2  = (const float*)d_in[11];
  const float* bc2  = (const float*)d_in[12];
  const float* Wv   = (const float*)d_in[13];
  const float* bv   = (const float*)d_in[14];
  const float* Woa  = (const float*)d_in[15];
  const float* boa  = (const float*)d_in[16];
  const float* g1   = (const float*)d_in[17];
  const float* be1  = (const float*)d_in[18];
  const float* Wm1  = (const float*)d_in[19];
  const float* bm1  = (const float*)d_in[20];
  const float* Wm2  = (const float*)d_in[21];
  const float* bm2  = (const float*)d_in[22];
  const float* g2   = (const float*)d_in[23];
  const float* be2  = (const float*)d_in[24];
  const float* Wcv  = (const float*)d_in[25];
  const float* bcv  = (const float*)d_in[26];
  const float* Wco  = (const float*)d_in[27];
  const float* bco  = (const float*)d_in[28];
  const float* scl  = (const float*)d_in[29];
  const float* Wp1  = (const float*)d_in[30];
  const float* bp1  = (const float*)d_in[31];
  const float* Wp2  = (const float*)d_in[32];
  const float* bp2  = (const float*)d_in[33];

  float* z  = (float*)d_out;            // [B,F]
  float* ld = z + (size_t)BQ * FQ;      // [B]

  char* base = (char*)d_ws; size_t cur = 0;
  auto alloc = [&](size_t bytes) -> void* {
    void* p = base + cur; cur += (bytes + 255) & ~(size_t)255; return p;
  };
  // bf16 weights
  bf16* bWi  = (bf16*)alloc((size_t)FQ * HQ * 2);
  bf16* bWc1 = (bf16*)alloc((size_t)CQ * 2 * HQ * 2);
  bf16* bWc2 = (bf16*)alloc((size_t)2 * HQ * HQ * 2);
  bf16* bWv  = (bf16*)alloc((size_t)LQ * HQ * HQ * 2);
  bf16* bWoa = (bf16*)alloc((size_t)LQ * HQ * HQ * 2);
  bf16* bWm1 = (bf16*)alloc((size_t)LQ * HQ * 4 * HQ * 2);
  bf16* bWm2 = (bf16*)alloc((size_t)LQ * 4 * HQ * HQ * 2);
  bf16* bWcv = (bf16*)alloc((size_t)LQ * HQ * HQ * 2);
  bf16* bWco = (bf16*)alloc((size_t)LQ * HQ * HQ * 2);
  bf16* bWp1 = (bf16*)alloc((size_t)HQ * HQ * 2);
  bf16* bWp2 = (bf16*)alloc((size_t)HQ * FQ * 2);
  // activations / scratch
  float* hb    = (float*)alloc((size_t)SQ * HQ * 4);         // per-step bias
  bf16*  cbf   = (bf16*)alloc((size_t)BQ * CQ * 2);
  bf16*  c1bf  = (bf16*)alloc((size_t)BQ * 2 * HQ * 2);
  bf16*  ctxbf = (bf16*)alloc((size_t)BQ * HQ * 2);
  bf16*  vctxb = (bf16*)alloc((size_t)BQ * HQ * 2);
  float* xatt  = (float*)alloc((size_t)LQ * BQ * HQ * 4);
  bf16*  zbf   = (bf16*)alloc((size_t)BQ * FQ * 2);
  bf16*  ubf   = (bf16*)alloc((size_t)BQ * FQ * 2);
  float* h     = (float*)alloc((size_t)BQ * HQ * 4);
  float* htg   = (float*)alloc((size_t)BQ * HQ * 4);
  bf16*  hbf   = (bf16*)alloc((size_t)BQ * HQ * 2);
  bf16*  htbf  = (bf16*)alloc((size_t)BQ * HQ * 2);
  bf16*  t1p   = (bf16*)alloc((size_t)BQ * HQ * 2);
  bf16*  t1t   = (bf16*)alloc((size_t)BQ * HQ * 2);
  float* dp    = (float*)alloc((size_t)BQ * HQ * 4);
  float* dtn   = (float*)alloc((size_t)BQ * HQ * 4);
  bf16*  mmp   = (bf16*)alloc((size_t)BQ * 4 * HQ * 2);
  bf16*  mmt   = (bf16*)alloc((size_t)BQ * 4 * HQ * 2);
  bf16*  gp    = (bf16*)alloc((size_t)BQ * HQ * 2);
  bf16*  gt    = (bf16*)alloc((size_t)BQ * HQ * 2);
  float* vb    = (float*)alloc((size_t)BQ * FQ * 4);
  float* jub   = (float*)alloc((size_t)BQ * FQ * 4);

  auto cast = [&](const float* s, bf16* d, int n) {
    castf2b<<<(n + 1023) / 1024, 256, 0, stream>>>(s, d, n);
  };
  auto gemm = [&](const bf16* Ap, const bf16* At, const bf16* Wb,
                  const float* bias, int N, int K, float* Yp32, float* Yt32,
                  bf16* Ypb, bf16* Ytb, int flags) {
    dim3 g(N / 64, BQ / 128);
    if (At) {
      switch (flags) {
        case 2: jvp_gemm<true, 2><<<g, 256, 0, stream>>>(Ap, At, Wb, bias, N, K, Yp32, Yt32, Ypb, Ytb); break;
        case 4: jvp_gemm<true, 4><<<g, 256, 0, stream>>>(Ap, At, Wb, bias, N, K, Yp32, Yt32, Ypb, Ytb); break;
        case 5: jvp_gemm<true, 5><<<g, 256, 0, stream>>>(Ap, At, Wb, bias, N, K, Yp32, Yt32, Ypb, Ytb); break;
        case 6: jvp_gemm<true, 6><<<g, 256, 0, stream>>>(Ap, At, Wb, bias, N, K, Yp32, Yt32, Ypb, Ytb); break;
      }
    } else {
      switch (flags) {
        case 2: jvp_gemm<false, 2><<<g, 256, 0, stream>>>(Ap, At, Wb, bias, N, K, Yp32, Yt32, Ypb, Ytb); break;
        case 4: jvp_gemm<false, 4><<<g, 256, 0, stream>>>(Ap, At, Wb, bias, N, K, Yp32, Yt32, Ypb, Ytb); break;
        case 5: jvp_gemm<false, 5><<<g, 256, 0, stream>>>(Ap, At, Wb, bias, N, K, Yp32, Yt32, Ypb, Ytb); break;
      }
    }
  };

  // ---- weight casts (once per call) ----
  cast(Wi,  bWi,  FQ * HQ);
  cast(Wc1, bWc1, CQ * 2 * HQ);
  cast(Wc2, bWc2, 2 * HQ * HQ);
  cast(Wv,  bWv,  LQ * HQ * HQ);
  cast(Woa, bWoa, LQ * HQ * HQ);
  cast(Wm1, bWm1, LQ * HQ * 4 * HQ);
  cast(Wm2, bWm2, LQ * 4 * HQ * HQ);
  cast(Wcv, bWcv, LQ * HQ * HQ);
  cast(Wco, bWco, LQ * HQ * HQ);
  cast(Wp1, bWp1, HQ * HQ);
  cast(Wp2, bWp2, HQ * FQ);

  // ---- precompute: time-embedding biases, context path, xatt[l] ----
  temb_kernel<<<SQ, HQ, 0, stream>>>(Wt1, bt1, Wt2, bt2, bi, hb);
  cast(ctxF, cbf, BQ * CQ);
  gemm(cbf,  nullptr, bWc1, bc1, 2 * HQ, CQ, nullptr, nullptr, c1bf,  nullptr, 1 | 4);
  gemm(c1bf, nullptr, bWc2, bc2, HQ, 2 * HQ, nullptr, nullptr, ctxbf, nullptr, 4);
  for (int l = 0; l < LQ; ++l) {
    gemm(ctxbf, nullptr, bWcv + (size_t)l * HQ * HQ, bcv + l * HQ, HQ, HQ,
         nullptr, nullptr, vctxb, nullptr, 4);
    gemm(vctxb, nullptr, bWco + (size_t)l * HQ * HQ, bco + l * HQ, HQ, HQ,
         xatt + (size_t)l * BQ * HQ, nullptr, nullptr, nullptr, 2);
  }
  init_zld<<<(BQ * FQ + 255) / 256, 256, 0, stream>>>(x, z, ld);

  // ---- 10 integration steps, each with full JVP ----
  for (int s = 0; s < SQ; ++s) {
    cast(z, zbf, BQ * FQ);
    cast(u + (size_t)s * BQ * FQ, ubf, BQ * FQ);
    // h = z@Wi + (bi + temb_s); ht = u@Wi
    gemm(zbf, ubf, bWi, hb + (size_t)s * HQ, HQ, FQ, h, htg, hbf, htbf, 2 | 4);
    for (int l = 0; l < LQ; ++l) {
      gemm(hbf, htbf, bWv + (size_t)l * HQ * HQ, bv + l * HQ, HQ, HQ,
           nullptr, nullptr, t1p, t1t, 4);
      gemm(t1p, t1t, bWoa + (size_t)l * HQ * HQ, boa + l * HQ, HQ, HQ,
           dp, dtn, nullptr, nullptr, 2);
      ln_jvp<<<BQ, 256, 0, stream>>>(h, htg, dp, dtn, g1 + l * HQ, be1 + l * HQ,
                                     nullptr, nullptr, 0, hbf, htbf);
      gemm(hbf, htbf, bWm1 + (size_t)l * HQ * 4 * HQ, bm1 + l * 4 * HQ, 4 * HQ, HQ,
           nullptr, nullptr, mmp, mmt, 1 | 4);
      gemm(mmp, mmt, bWm2 + (size_t)l * 4 * HQ * HQ, bm2 + l * HQ, HQ, 4 * HQ,
           dp, dtn, nullptr, nullptr, 2);
      ln_jvp<<<BQ, 256, 0, stream>>>(h, htg, dp, dtn, g2 + l * HQ, be2 + l * HQ,
                                     xatt + (size_t)l * BQ * HQ, scl, l, hbf, htbf);
    }
    gemm(hbf, htbf, bWp1, bp1, HQ, HQ, nullptr, nullptr, gp, gt, 1 | 4);
    gemm(gp, gt, bWp2, bp2, FQ, HQ, vb, jub, nullptr, nullptr, 2);
    update_zld<<<BQ / 8, 256, 0, stream>>>(z, ld, vb, jub,
                                           u + (size_t)s * BQ * FQ, 0.1f);
  }
}
